// My_LSTM_cell_89773406421453
// MI455X (gfx1250) — matrix operand order repaired and checked by reading, not compiled, not run on hardware
//
#include <hip/hip_runtime.h>
#include <hip/hip_bf16.h>

// ---------------------------------------------------------------------------
// LSTM cell on MI455X (gfx1250): bf16 WMMA GEMM fused with gate nonlinearity.
//
//   gates[B,4H] = x[B,K] * W^T[4H,K] + h[B,K] * R^T[4H,K] + bias
//   c_next = sigm(gf)*c_prev + sigm(gi)*tanh(gc) ; h_next = sigm(go)*tanh(c_next)
//
// Phase 1: f32 -> bf16 (RNE) into d_ws.
// Phase 2: fused WMMA kernel. Each block owns a 256(batch) x 32(hidden) tile
// and accumulates ALL FOUR gates. Weight tiles (shared by all 8 waves) are
// staged through double-buffered LDS with a software pipeline:
//     global prefetch (regs) -> wmma from LDS[cur] -> ds_store LDS[next] -> barrier
// Each wave owns TWO 16-row batch sub-tiles, so every 1KB B fragment pulled
// from LDS feeds 2 WMMAs (512 B/WMMA of DS traffic; A is 128 B/WMMA of VMEM).
// Epilogue uses __expf-based sigmoid/tanh (v_exp_f32) instead of libm tanhf.
// ---------------------------------------------------------------------------

#define BATCH   16384
#define DIM     1024      // IN_LEN == HID_LEN
#define GATES   4
#define BM      256       // batch rows per block (32 per wave, 8 waves)
#define MSUB    2         // 16-row sub-tiles per wave
#define BN      32        // hidden cols per block (2 WMMA tiles per wave)
#define BK      32        // K step (bf16 WMMA depth)
#define NT      (DIM / BK)        // 32 K-iterations
#define TROWS   (GATES * BN)      // 128 weight rows per tile (4 gates x 32 cols)
#define RSTR    40                // LDS row stride in ushorts: 32 data + 8 pad (80B)
                                  //  -> 16-row windows hit distinct bank groups,
                                  //     every b128 access stays 16B-aligned.

typedef __attribute__((ext_vector_type(16))) __bf16          v16bf;
typedef __attribute__((ext_vector_type(8)))  float           v8f;
typedef __attribute__((ext_vector_type(4)))  float           v4f;
typedef __attribute__((ext_vector_type(8)))  unsigned short  us8;   // 16 bytes

union Frag16 {             // 32 bytes: one bf16 WMMA A/B operand per lane
    v16bf v;
    us8   h[2];
};

// ---- f32 -> bf16, round-to-nearest-even ---------------------------------
__device__ __forceinline__ unsigned short f32_to_bf16(float f) {
    unsigned int u = __float_as_uint(f);
    u += 0x7FFFu + ((u >> 16) & 1u);
    return (unsigned short)(u >> 16);
}

__global__ __launch_bounds__(256)
void cvt_f32_to_bf16(const float* __restrict__ src,
                     unsigned short* __restrict__ dst, int n) {
    int i = (blockIdx.x * 256 + threadIdx.x) * 8;   // 8 elems/thread
    if (i >= n) return;
    v4f a = *(const v4f*)(src + i);
    v4f b = *(const v4f*)(src + i + 4);
    us8 o;
    o[0] = f32_to_bf16(a[0]); o[1] = f32_to_bf16(a[1]);
    o[2] = f32_to_bf16(a[2]); o[3] = f32_to_bf16(a[3]);
    o[4] = f32_to_bf16(b[0]); o[5] = f32_to_bf16(b[1]);
    o[6] = f32_to_bf16(b[2]); o[7] = f32_to_bf16(b[3]);
    *(us8*)(dst + i) = o;                            // global_store_b128
}

// ---- WMMA A operand fetch (CDNA5 ISA 7.12.2, wave32) ---------------
// A (16x32 bf16, row-major source, leading dim ld):
//   lane L: row = L&15, half = L>>4
//   V0..3 = K[half*8 .. +7], V4..7 = K[16+half*8 .. +7]
__device__ __forceinline__ v16bf
load_a_frag(const unsigned short* __restrict__ base, int ld,
            int row0, int k0, int lane) {
    const int r  = lane & 15;
    const int hi = lane >> 4;
    const unsigned short* p = base + (size_t)(row0 + r) * ld + k0 + hi * 8;
    Frag16 f;
    f.h[0] = *(const us8*)(p);        // K = k0+hi*8    .. +7
    f.h[1] = *(const us8*)(p + 16);   // K = k0+16+hi*8 .. +7
    return f.v;
}

// ---- fast transcendentals: straight-line v_exp_f32 paths ----------------
__device__ __forceinline__ float sigmoidf_fast(float x) {
    return 1.0f / (1.0f + __expf(-x));           // v_exp + v_rcp class
}
__device__ __forceinline__ float tanhf_fast(float x) {
    // tanh(x) = 1 - 2/(1 + e^{2x});  monotone & saturating, no branches.
    return 1.0f - 2.0f / (1.0f + __expf(2.0f * x));
}

// ---- fused LSTM WMMA kernel ---------------------------------------------
__global__ __launch_bounds__(256)
void lstm_wmma_fused(const unsigned short* __restrict__ xb,   // [BATCH, DIM] bf16
                     const unsigned short* __restrict__ hb,   // [BATCH, DIM] bf16
                     const unsigned short* __restrict__ Wb,   // [4*DIM, DIM] bf16
                     const unsigned short* __restrict__ Rb,   // [4*DIM, DIM] bf16
                     const float* __restrict__ b1, const float* __restrict__ b2,
                     const float* __restrict__ b3, const float* __restrict__ b4,
                     const float* __restrict__ c_prev,        // [BATCH, DIM]
                     float* __restrict__ h_out,               // [BATCH, DIM]
                     float* __restrict__ c_out) {             // [BATCH, DIM]
    // Double-buffered weight tiles: [buf][W|R][TROWS * RSTR] bf16  (40 KB)
    __shared__ unsigned short smem[2][2][TROWS * RSTR];

    const int tid   = threadIdx.x;
    const int bid   = blockIdx.x;
    const int tileN = bid & (DIM / BN - 1);     // 32 hidden tiles
    const int tileM = bid >> 5;                 // 64 batch tiles
    const int wave  = tid >> 5;
    const int lane  = tid & 31;
    const int m0    = tileM * BM + wave * (16 * MSUB);  // wave's 32 batch rows
    const int n0    = tileN * BN;               // 32 hidden cols (2 tiles of 16)
    const int nl    = lane & 15;
    const int hi    = lane >> 4;

    // ---- cooperative-fill mapping: 256 threads x 64B (32B of W, 32B of R)
    const int    fr   = tid >> 1;                               // tile row 0..127
    const int    fh   = tid & 1;                                // 16-elem half
    const int    fg   = fr >> 5;                                // gate
    const int    fc   = fr & 31;                                // col within tile
    const size_t gsrc = (size_t)(fg * DIM + n0 + fc) * DIM;     // W/R row start
    const int    fko  = fh * 16;                                // elem offset in K
    const int    ldsb = fr * RSTR + fko;                        // elem offset in LDS

    // ---- preload K-tile 0 into buffer 0 ----
    {
        const us8* sw = (const us8*)(Wb + gsrc + fko);
        const us8* sr = (const us8*)(Rb + gsrc + fko);
        us8 w0 = sw[0], w1 = sw[1], r0 = sr[0], r1 = sr[1];
        *(us8*)&smem[0][0][ldsb]     = w0;
        *(us8*)&smem[0][0][ldsb + 8] = w1;
        *(us8*)&smem[0][1][ldsb]     = r0;
        *(us8*)&smem[0][1][ldsb + 8] = r1;
    }
    __syncthreads();

    v8f acc[GATES][2][MSUB];
#pragma unroll
    for (int g = 0; g < GATES; ++g)
#pragma unroll
        for (int j = 0; j < 2; ++j)
#pragma unroll
            for (int ms = 0; ms < MSUB; ++ms)
                acc[g][j][ms] = (v8f)(0.0f);

    // ---- pipelined K loop: 32 WMMAs / iteration / wave ----
    for (int t = 0; t < NT; ++t) {
        const int  cur      = t & 1;
        const bool has_next = (t + 1) < NT;

        // 1) issue global prefetch of K-tile t+1 (weights) into registers
        us8 pw0, pw1, pr0, pr1;
        if (has_next) {
            const int  kk = (t + 1) * BK;
            const us8* sw = (const us8*)(Wb + gsrc + kk + fko);
            const us8* sr = (const us8*)(Rb + gsrc + kk + fko);
            pw0 = sw[0]; pw1 = sw[1];
            pr0 = sr[0]; pr1 = sr[1];
        }

        // 2) A fragments for the current K-tile: 2 batch sub-tiles of x and h
        const int   k0  = t * BK;
        const v16bf ax0 = load_a_frag(xb, DIM, m0,      k0, lane);
        const v16bf ax1 = load_a_frag(xb, DIM, m0 + 16, k0, lane);
        const v16bf ah0 = load_a_frag(hb, DIM, m0,      k0, lane);
        const v16bf ah1 = load_a_frag(hb, DIM, m0 + 16, k0, lane);

        // 3) 32 WMMAs fed from LDS; each B fragment feeds 2 WMMAs
#pragma unroll
        for (int g = 0; g < GATES; ++g) {
#pragma unroll
            for (int j = 0; j < 2; ++j) {
                const int row = g * BN + j * 16 + nl;     // B lane: col = lane&15
                const int off = row * RSTR + hi * 8;      // K = hi*8..+7, 16+hi*8..+7
                Frag16 fw, fr2;
                fw.h[0]  = *(const us8*)&smem[cur][0][off];
                fw.h[1]  = *(const us8*)&smem[cur][0][off + 16];
                fr2.h[0] = *(const us8*)&smem[cur][1][off];
                fr2.h[1] = *(const us8*)&smem[cur][1][off + 16];
                acc[g][j][0] = __builtin_amdgcn_wmma_f32_16x16x32_bf16(
                    false, ax0, false, fw.v, (short)0, acc[g][j][0], false, false);
                acc[g][j][1] = __builtin_amdgcn_wmma_f32_16x16x32_bf16(
                    false, ax1, false, fw.v, (short)0, acc[g][j][1], false, false);
                acc[g][j][0] = __builtin_amdgcn_wmma_f32_16x16x32_bf16(
                    false, ah0, false, fr2.v, (short)0, acc[g][j][0], false, false);
                acc[g][j][1] = __builtin_amdgcn_wmma_f32_16x16x32_bf16(
                    false, ah1, false, fr2.v, (short)0, acc[g][j][1], false, false);
            }
        }

        // 4) commit prefetched tile into the other buffer (safe: that buffer
        //    was last read in iteration t-1, behind the previous barrier)
        if (has_next) {
            const int nxt = cur ^ 1;
            *(us8*)&smem[nxt][0][ldsb]     = pw0;
            *(us8*)&smem[nxt][0][ldsb + 8] = pw1;
            *(us8*)&smem[nxt][1][ldsb]     = pr0;
            *(us8*)&smem[nxt][1][ldsb + 8] = pr1;
        }
        __syncthreads();
    }

    // ---- epilogue: C/D layout: VGPR r, lane L -> M = r + 8*(L>>4), N = L&15
#pragma unroll
    for (int j = 0; j < 2; ++j) {
        const int n  = n0 + j * 16 + nl;
        const float bf = b1[n];
        const float bi = b2[n];
        const float bc = b3[n];
        const float bo = b4[n];
#pragma unroll
        for (int ms = 0; ms < MSUB; ++ms) {
#pragma unroll
            for (int r = 0; r < 8; ++r) {
                const int    m   = m0 + ms * 16 + r + 8 * hi;
                const size_t idx = (size_t)m * DIM + n;
                const float f  = sigmoidf_fast(acc[0][j][ms][r] + bf);
                const float i  = sigmoidf_fast(acc[1][j][ms][r] + bi);
                const float g  = tanhf_fast(acc[2][j][ms][r] + bc);
                const float o  = sigmoidf_fast(acc[3][j][ms][r] + bo);
                const float cn = f * c_prev[idx] + i * g;
                h_out[idx] = o * tanhf_fast(cn);
                c_out[idx] = cn;
            }
        }
    }
}

// ---------------------------------------------------------------------------
extern "C" void kernel_launch(void* const* d_in, const int* in_sizes, int n_in,
                              void* d_out, int out_size, void* d_ws, size_t ws_size,
                              hipStream_t stream) {
    // Input order: x, h, c_prev, then (w,b,r) per gate.
    const float* x      = (const float*)d_in[0];
    const float* h      = (const float*)d_in[1];
    const float* c_prev = (const float*)d_in[2];
    const float* w[4]   = { (const float*)d_in[3], (const float*)d_in[6],
                            (const float*)d_in[9], (const float*)d_in[12] };
    const float* b[4]   = { (const float*)d_in[4], (const float*)d_in[7],
                            (const float*)d_in[10], (const float*)d_in[13] };
    const float* r[4]   = { (const float*)d_in[5], (const float*)d_in[8],
                            (const float*)d_in[11], (const float*)d_in[14] };

    // Workspace layout (bf16 as ushort): xb, hb, Wb[4H,K], Rb[4H,K]  (~84 MB)
    const size_t ACT  = (size_t)BATCH * DIM;      // 16,777,216
    const size_t WMAT = (size_t)DIM * DIM;        //  1,048,576 per gate
    unsigned short* xb = (unsigned short*)d_ws;
    unsigned short* hb = xb + ACT;
    unsigned short* Wb = hb + ACT;
    unsigned short* Rb = Wb + GATES * WMAT;

    // ---- Phase 1: f32 -> bf16 conversions (8 elems/thread, b128 I/O) ----
    const int CVT_TPB = 256, CVT_EPT = 8;
    {
        int n = (int)ACT;
        int grid = n / (CVT_TPB * CVT_EPT);
        cvt_f32_to_bf16<<<grid, CVT_TPB, 0, stream>>>(x, xb, n);
        cvt_f32_to_bf16<<<grid, CVT_TPB, 0, stream>>>(h, hb, n);
    }
    {
        int n = (int)WMAT;
        int grid = n / (CVT_TPB * CVT_EPT);
        for (int g = 0; g < GATES; ++g) {
            cvt_f32_to_bf16<<<grid, CVT_TPB, 0, stream>>>(w[g], Wb + g * WMAT, n);
            cvt_f32_to_bf16<<<grid, CVT_TPB, 0, stream>>>(r[g], Rb + g * WMAT, n);
        }
    }

    // ---- Phase 2: fused WMMA GEMM + LSTM nonlinearity -------------------
    float* h_out = (float*)d_out;                 // [BATCH, DIM]
    float* c_out = h_out + ACT;                   // [BATCH, DIM]
    const int grid = (BATCH / BM) * (DIM / BN);   // 64 * 32 = 2048 blocks
    lstm_wmma_fused<<<grid, 256, 0, stream>>>(xb, hb, Wb, Rb,
                                              b[0], b[1], b[2], b[3],
                                              c_prev, h_out, c_out);
}
